// SGMTSEncoder_81415400063703
// MI455X (gfx1250) — matrix operand order repaired and checked by reading, not compiled, not run on hardware
//
#include <hip/hip_runtime.h>
#include <hip/hip_bf16.h>

// ---------------------------------------------------------------------------
// CDNA5 (gfx1250) implementation of SGMTSEncoder.
//  - conv2 (77 GFLOP, dominant) -> implicit GEMM via v_wmma_f32_16x16x32_f16
//  - fc / delta-B-C projection / out projection -> generic WMMA GEMM
//  - MST+BFS structure -> single-block serial per batch (stays on-stream)
//  - tree scan -> one thread per DF channel, no barriers needed
// B operand staged N-major in LDS (fragment loads = ds_load_b128 pairs);
// this revision also vectorizes the LDS staging (v8h/v4h copies, batched
// im2col gathers) to kill the per-element global_load_u16 -> wait -> b16 chains.
// ---------------------------------------------------------------------------

typedef __attribute__((ext_vector_type(16))) _Float16 v16h;
typedef __attribute__((ext_vector_type(8)))  _Float16 v8h;
typedef __attribute__((ext_vector_type(4)))  _Float16 v4h;
typedef __attribute__((ext_vector_type(8)))  float    v8f;

// ---- WMMA fragment helpers (layouts from CDNA5 ISA 7.12.2) -----

// A-matrix 16x32 f16, row-major in LDS with leading dim `ld` (halves).
// lane's data: row (lane&15), halves [kb..kb+7] and [kb+16..kb+23],
// kb = (lane>=16 ? 8 : 0). Both segments are 16B-aligned if ld%8==0.
__device__ __forceinline__ v16h load_a_frag(const _Float16* base, int ld, int lane) {
  int m  = lane & 15;
  int kb = (lane >> 4) << 3;
  const _Float16* row = base + m * ld + kb;
  v8h lo = *(const v8h*)(row);
  v8h hi = *(const v8h*)(row + 16);
  return __builtin_shufflevector(lo, hi, 0, 1, 2, 3, 4, 5, 6, 7,
                                 8, 9, 10, 11, 12, 13, 14, 15);
}

// B-matrix 32x16 f16 staged N-MAJOR in LDS: Bl[n][k], leading dim `ld`.
// lane's data: row (lane&15), halves [kb..kb+7] and [kb+16..kb+23], kb = (lane>=16 ? 8 : 0).
// Fully contiguous -> two ds_load_b128.
__device__ __forceinline__ v16h load_bT_frag(const _Float16* base, int ld, int lane) {
  int n  = lane & 15;
  int kb = (lane >> 4) << 3;
  const _Float16* row = base + n * ld + kb;
  v8h lo = *(const v8h*)(row);
  v8h hi = *(const v8h*)(row + 16);
  return __builtin_shufflevector(lo, hi, 0, 1, 2, 3, 4, 5, 6, 7,
                                 8, 9, 10, 11, 12, 13, 14, 15);
}

__device__ __forceinline__ v8f wmma_f16(v16h a, v16h b, v8f c) {
  return __builtin_amdgcn_wmma_f32_16x16x32_f16(false, a, false, b, (short)0, c, false, false);
}

__device__ __forceinline__ float gelu_exact(float x) {
  return 0.5f * x * (1.0f + erff(x * 0.70710678118654752f));
}

// ---------------------------------------------------------------------------
// 0) Weight preparation: reorder / transpose / f16-convert for WMMA B-layout.
// ---------------------------------------------------------------------------
__global__ void prep_weights_kernel(const float* __restrict__ c2w,
                                    const float* __restrict__ fcw,
                                    const float* __restrict__ Wdw,
                                    const float* __restrict__ Bw,
                                    const float* __restrict__ Cw,
                                    const float* __restrict__ outw,
                                    const float* __restrict__ Wdb,
                                    const float* __restrict__ Bb,
                                    const float* __restrict__ Cb,
                                    _Float16* __restrict__ w2r,
                                    _Float16* __restrict__ fcB,
                                    _Float16* __restrict__ WALL,
                                    _Float16* __restrict__ WoutB,
                                    float* __restrict__ biasALL) {
  int stride = gridDim.x * blockDim.x;
  int tid0 = blockIdx.x * blockDim.x + threadIdx.x;
  // conv2 weights: [oc][k] with k = tap*64 + ic  (tap = kh*3+kw)
  for (int i = tid0; i < 128 * 576; i += stride) {
    int oc = i / 576, k = i % 576;
    int tap = k >> 6, ic = k & 63;
    w2r[oc * 576 + k] = (_Float16)c2w[oc * 576 + ic * 9 + tap];
  }
  // fc_w (256,128) -> B[k][n] = fc_w[n][k]
  for (int i = tid0; i < 128 * 256; i += stride) {
    int k = i >> 8, n = i & 255;
    fcB[k * 256 + n] = (_Float16)fcw[n * 128 + k];
  }
  // fused [Wd_w^T | B_w^T | C_w^T] : 256 x 288
  for (int i = tid0; i < 256 * 288; i += stride) {
    int k = i / 288, n = i % 288;
    float v;
    if (n < 256)      v = Wdw[n * 256 + k];
    else if (n < 272) v = Bw[(n - 256) * 256 + k];
    else              v = Cw[(n - 272) * 256 + k];
    WALL[k * 288 + n] = (_Float16)v;
  }
  // out_w (256,256) -> transpose
  for (int i = tid0; i < 256 * 256; i += stride) {
    int k = i >> 8, n = i & 255;
    WoutB[k * 256 + n] = (_Float16)outw[n * 256 + k];
  }
  for (int i = tid0; i < 288; i += stride)
    biasALL[i] = (i < 256) ? Wdb[i] : (i < 272 ? Bb[i - 256] : Cb[i - 272]);
}

// ---------------------------------------------------------------------------
// 1) conv1 (3->64, 3x3, SAME within 16x16 patch) + GELU, output f16.
// ---------------------------------------------------------------------------
__global__ __launch_bounds__(256) void conv1_kernel(const float* __restrict__ images,
                                                    const float* __restrict__ w,
                                                    const float* __restrict__ bias,
                                                    _Float16* __restrict__ out) {
  __shared__ float patch[3][18][18];   // zero-padded halo
  __shared__ float wc[64 * 27];
  int pid = blockIdx.x;
  int b = pid >> 10, pp = pid & 1023;
  int ph = pp >> 5, pw = pp & 31;
  int tid = threadIdx.x;

  for (int i = tid; i < 3 * 18 * 18; i += 256) (&patch[0][0][0])[i] = 0.f;
  for (int i = tid; i < 1728; i += 256) wc[i] = w[i];
  __syncthreads();
  for (int i = tid; i < 768; i += 256) {
    int c = i >> 8, yx = i & 255, y = yx >> 4, x = yx & 15;
    patch[c][y + 1][x + 1] =
        images[(((long)b * 3 + c) * 512 + ph * 16 + y) * 512 + pw * 16 + x];
  }
  __syncthreads();
  int y = tid >> 4, x = tid & 15;
  for (int oc = 0; oc < 64; ++oc) {
    float acc = bias[oc];
#pragma unroll
    for (int c = 0; c < 3; ++c)
#pragma unroll
      for (int kh = 0; kh < 3; ++kh)
#pragma unroll
        for (int kw = 0; kw < 3; ++kw)
          acc += patch[c][y + kh][x + kw] * wc[oc * 27 + c * 9 + kh * 3 + kw];
    out[((long)pid * 64 + oc) * 256 + tid] = (_Float16)gelu_exact(acc);
  }
}

// ---------------------------------------------------------------------------
// 2) conv2 (64->128, 3x3) as implicit GEMM with WMMA, fused GELU + mean pool.
//    M=128 (oc), N=256 (pixels), K=576. One block (512 thr / 16 waves) per
//    patch; wave w owns M-tile (w&7) and 8 N-tiles of half the pixel range.
//    Vectorized staging: A = one v8h copy/thread; B = batched 8-wide im2col
//    gathers + ds_store_b128 into the N-major tile.
// ---------------------------------------------------------------------------
__global__ __launch_bounds__(512) void conv2_pool_kernel(const _Float16* __restrict__ conv1out,
                                                         const _Float16* __restrict__ w2r,
                                                         const float* __restrict__ bias2,
                                                         float* __restrict__ pooled,
                                                         _Float16* __restrict__ pooledh) {
  __shared__ __align__(16) _Float16 Al[128][40];   // 128 oc x 32 K (+pad)
  __shared__ __align__(16) _Float16 Bl[256][40];   // 256 pixels x 32 K (+pad)
  __shared__ float pool_lds[128];

  int pid = blockIdx.x;
  const _Float16* src = conv1out + (long)pid * 16384;
  int tid = threadIdx.x;
  if (tid < 128) pool_lds[tid] = 0.f;
  int wv = tid >> 5, lane = tid & 31;
  int mt = wv & 7, ng = wv >> 3;

  v8f zero = {0.f, 0.f, 0.f, 0.f, 0.f, 0.f, 0.f, 0.f};
  v8f acc[8];
#pragma unroll
  for (int j = 0; j < 8; ++j) acc[j] = zero;

  // A-staging coordinates (constant across K loop): one 16B copy per thread
  int ar = tid >> 2, ac8 = (tid & 3) * 8;

  for (int kc = 0; kc < 18; ++kc) {        // K = 576 = 18 * 32
    __syncthreads();
    // A chunk 128x32: 4096 halves / 512 threads = one v8h each (aligned)
    {
      v8h v = *(const v8h*)(w2r + ar * 576 + kc * 32 + ac8);
      *(v8h*)&Al[ar][ac8] = v;
    }
    // B chunk (im2col, N-major): 2 jobs x 8 consecutive kk per thread.
    int tap = kc >> 1, kh = tap / 3, kw = tap % 3;   // k = tap*64 + ic
    int icbase = (kc & 1) * 32;
#pragma unroll
    for (int q = 0; q < 2; ++q) {
      int j = tid + 512 * q;               // 1024 jobs of 8 halves
      int pix = j & 255, kg = (j >> 8) * 8;
      int y = pix >> 4, x = pix & 15;
      int yy = y + kh - 1, xx = x + kw - 1;
      v8h v;
      if (yy >= 0 && yy < 16 && xx >= 0 && xx < 16) {
        const _Float16* s = src + (icbase + kg) * 256 + yy * 16 + xx;
#pragma unroll
        for (int e = 0; e < 8; ++e) v[e] = s[e * 256];
      } else {
#pragma unroll
        for (int e = 0; e < 8; ++e) v[e] = (_Float16)0.f;
      }
      *(v8h*)&Bl[pix][kg] = v;             // ds_store_b128
    }
    __syncthreads();
    v16h a = load_a_frag(&Al[mt * 16][0], 40, lane);
#pragma unroll
    for (int j = 0; j < 8; ++j) {
      v16h bf = load_bT_frag(&Bl[(ng * 8 + j) * 16][0], 40, lane);
      acc[j] = wmma_f16(a, bf, acc[j]);
    }
  }
  __syncthreads();
  // bias + GELU + mean-pool over pixels (N dimension)
  int mloc = (lane >> 4) << 3;
#pragma unroll
  for (int r = 0; r < 8; ++r) {
    int oc = mt * 16 + mloc + r;
    float bb = bias2[oc];
    float s = 0.f;
#pragma unroll
    for (int j = 0; j < 8; ++j) s += gelu_exact(acc[j][r] + bb);
    atomicAdd(&pool_lds[oc], s);
  }
  __syncthreads();
  if (tid < 128) {
    float v = pool_lds[tid] * (1.f / 256.f);
    pooled[pid * 128 + tid] = v;
    pooledh[pid * 128 + tid] = (_Float16)v;
  }
}

// ---------------------------------------------------------------------------
// 3) Generic WMMA GEMM: C[M,N] = A[M,K](f16,row) * B[K,N](f16,row) + bias.
//    Block tile 32x128 (8 waves in 2x4), K step 32. B staged N-major in LDS.
// ---------------------------------------------------------------------------
__global__ __launch_bounds__(256) void gemm_f16_kernel(const _Float16* __restrict__ A,
                                                       const _Float16* __restrict__ B,
                                                       const float* __restrict__ bias,
                                                       float* __restrict__ C,
                                                       _Float16* __restrict__ Ch,
                                                       int M, int N, int K) {
  __shared__ __align__(16) _Float16 Al[32][40];
  __shared__ __align__(16) _Float16 Bl[128][40];    // [n][k]
  int bm = blockIdx.x * 32;
  int bn = blockIdx.y * 128;
  int tid = threadIdx.x, wv = tid >> 5, lane = tid & 31;
  int wm = wv & 1, wn = wv >> 1;

  v8f acc0 = {0.f, 0.f, 0.f, 0.f, 0.f, 0.f, 0.f, 0.f};
  v8f acc1 = acc0;

  int ar = tid >> 3, ac4 = (tid & 7) * 4;           // A stage: one v4h each

  for (int kc = 0; kc < K; kc += 32) {
    __syncthreads();
    if (kc + 32 < K)  // hint next B chunk into cache (global_prefetch_b8)
      __builtin_prefetch(B + (long)(kc + 32) * N + bn + (tid & 127), 0, 0);
    // A chunk 32x32: 1024 halves / 256 threads = one v4h each (8B aligned)
    {
      int gm = bm + ar;
      v4h v;
      if (gm < M) v = *(const v4h*)(A + (long)gm * K + kc + ac4);
      else { v[0] = v[1] = v[2] = v[3] = (_Float16)0.f; }
      *(v4h*)&Al[ar][ac4] = v;
    }
    // B chunk 32x128 -> Bl[n][k]: 2 jobs x 8 consecutive k per thread.
#pragma unroll
    for (int q = 0; q < 2; ++q) {
      int j = tid + 256 * q;               // 512 jobs of 8 halves
      int n = j & 127, kg = (j >> 7) * 8;
      int gn = bn + n;
      v8h v;
      if (gn < N) {
#pragma unroll
        for (int e = 0; e < 8; ++e) v[e] = B[(long)(kc + kg + e) * N + gn];
      } else {
#pragma unroll
        for (int e = 0; e < 8; ++e) v[e] = (_Float16)0.f;
      }
      *(v8h*)&Bl[n][kg] = v;               // ds_store_b128
    }
    __syncthreads();
    v16h a  = load_a_frag(&Al[wm * 16][0], 40, lane);
    v16h b0 = load_bT_frag(&Bl[wn * 32][0], 40, lane);
    v16h b1 = load_bT_frag(&Bl[wn * 32 + 16][0], 40, lane);
    acc0 = wmma_f16(a, b0, acc0);
    acc1 = wmma_f16(a, b1, acc1);
  }
  int mloc = (lane >> 4) << 3;
  int ncol = lane & 15;
#pragma unroll
  for (int r = 0; r < 8; ++r) {
    int gm = bm + wm * 16 + mloc + r;
    if (gm >= M) continue;
    int gn0 = bn + wn * 32 + ncol;
    int gn1 = gn0 + 16;
    if (gn0 < N) {
      float v = acc0[r] + (bias ? bias[gn0] : 0.f);
      C[(long)gm * N + gn0] = v;
      if (Ch) Ch[(long)gm * N + gn0] = (_Float16)v;
    }
    if (gn1 < N) {
      float v = acc1[r] + (bias ? bias[gn1] : 0.f);
      C[(long)gm * N + gn1] = v;
      if (Ch) Ch[(long)gm * N + gn1] = (_Float16)v;
    }
  }
}

// ---------------------------------------------------------------------------
// 4) Language projections: gproj = lang @ W_gate^T, gprime = lang @ W_g'^T.
// ---------------------------------------------------------------------------
__global__ void lang_proj_kernel(const float* __restrict__ lang,
                                 const float* __restrict__ Wg,
                                 const float* __restrict__ Wgp,
                                 float* __restrict__ gproj,
                                 float* __restrict__ gprime) {
  int b = blockIdx.x, d = threadIdx.x;  // 2 x 256
  const float* L = lang + b * 896;
  float s0 = 0.f, s1 = 0.f;
  for (int k = 0; k < 896; ++k) {
    float lv = L[k];
    s0 += lv * Wg[d * 896 + k];
    s1 += lv * Wgp[d * 896 + k];
  }
  gproj[b * 256 + d] = s0;
  gprime[b * 256 + d] = s1;
}

// 5) r_sem = sigmoid(<gproj_b, feats_bp> / 16)
__global__ void rsem_kernel(const float* __restrict__ gproj,
                            const float* __restrict__ feats,
                            float* __restrict__ r) {
  int i = blockIdx.x * blockDim.x + threadIdx.x;  // 2048
  int b = i >> 10;
  const float* g = gproj + b * 256;
  const float* f = feats + (long)i * 256;
  float s = 0.f;
  for (int k = 0; k < 256; ++k) s += g[k] * f[k];
  s *= 0.0625f;  // 1/sqrt(256)
  r[i] = 1.f / (1.f + expf(-s));
}

// ---------------------------------------------------------------------------
// 6) Structure: Kruskal MST over 32x32 grid + BFS ordering. One block per
//    batch; edge weights in parallel, MST/BFS serial on lane 0 (tiny).
// ---------------------------------------------------------------------------
__global__ __launch_bounds__(256) void structure_kernel(const float* __restrict__ feats,
                                                        const float* __restrict__ r,
                                                        int* __restrict__ bfs,
                                                        int* __restrict__ par) {
  __shared__ float invn[1024];
  __shared__ float ew[1984];
  __shared__ int parent[1024];
  __shared__ int head[1024], tail[1024];
  __shared__ int nxt[2048], eto[2048];
  __shared__ int parof[1024];
  __shared__ int o2b[1024];
  __shared__ unsigned char vis[1024];
  int b = blockIdx.x, tid = threadIdx.x;
  const float* F = feats + (long)b * 1024 * 256;
  const float* R = r + b * 1024;

  for (int p = tid; p < 1024; p += 256) {
    const float* f = F + p * 256;
    float s = 0.f;
    for (int k = 0; k < 256; ++k) s += f[k] * f[k];
    invn[p] = 1.f / fmaxf(sqrtf(s), 1e-12f);
    head[p] = -1; tail[p] = -1; parent[p] = p;
    vis[p] = 0; parof[p] = -1;
  }
  __syncthreads();
  for (int e = tid; e < 1984; e += 256) {
    int u, v;
    if (e < 992) { int row = e / 31, col = e % 31; u = row * 32 + col; v = u + 1; }
    else         { int e2 = e - 992; u = (e2 >> 5) * 32 + (e2 & 31); v = u + 32; }
    const float* fu = F + u * 256;
    const float* fv = F + v * 256;
    float s = 0.f;
    for (int k = 0; k < 256; ++k) s += fu[k] * fv[k];
    float cosv = s * invn[u] * invn[v];
    ew[e] = (1.f - R[u]) * (1.f - R[v]) * (-cosv) + 1e-6f;
  }
  __syncthreads();
  if (tid == 0) {
    int cnt = 0, ne = 0;
    while (cnt < 1023) {
      float best = 3.2e38f; int bi = -1;
      for (int e = 0; e < 1984; ++e)
        if (ew[e] < best) { best = ew[e]; bi = e; }
      if (bi < 0) break;
      ew[bi] = __builtin_inff();
      int u, v;
      if (bi < 992) { int row = bi / 31, col = bi % 31; u = row * 32 + col; v = u + 1; }
      else          { int e2 = bi - 992; u = (e2 >> 5) * 32 + (e2 & 31); v = u + 32; }
      int ru = u; while (parent[ru] != ru) { parent[ru] = parent[parent[ru]]; ru = parent[ru]; }
      int rv = v; while (parent[rv] != rv) { parent[rv] = parent[parent[rv]]; rv = parent[rv]; }
      if (ru != rv) {
        parent[ru] = rv;
        nxt[ne] = -1; eto[ne] = v;
        if (head[u] < 0) head[u] = ne; else nxt[tail[u]] = ne;
        tail[u] = ne; ++ne;
        nxt[ne] = -1; eto[ne] = u;
        if (head[v] < 0) head[v] = ne; else nxt[tail[v]] = ne;
        tail[v] = ne; ++ne;
        ++cnt;
      }
    }
    int root = 0; float best = R[0];
    for (int p = 1; p < 1024; ++p) if (R[p] > best) { best = R[p]; root = p; }
    int* q = bfs + b * 1024;
    int qh = 0, qt = 0;
    q[qt++] = root; vis[root] = 1;
    while (qh < qt) {
      int n = q[qh++];
      for (int e = head[n]; e >= 0; e = nxt[e]) {
        int nb = eto[e];
        if (!vis[nb]) { vis[nb] = 1; parof[nb] = n; q[qt++] = nb; }
      }
    }
    for (int i = 0; i < 1024; ++i) o2b[q[i]] = i;
    for (int i = 0; i < 1024; ++i) {
      int po = parof[q[i]];
      par[b * 1024 + i] = (po < 0) ? -1 : o2b[po];
    }
  }
}

// 7) X = feats[bfs] + r[bfs]*g'  (f32 + f16 copies) ; rs = r[bfs]
__global__ void build_x_kernel(const float* __restrict__ feats,
                               const float* __restrict__ r,
                               const float* __restrict__ gprime,
                               const int* __restrict__ bfs,
                               float* __restrict__ Xf,
                               _Float16* __restrict__ Xh,
                               float* __restrict__ rs) {
  long i = blockIdx.x;  // 2048 (BFS-order row)
  int t = threadIdx.x;
  int b = (int)(i >> 10);
  int node = bfs[i];
  float rv = r[b * 1024 + node];
  float x = feats[((long)b * 1024 + node) * 256 + t] + rv * gprime[b * 256 + t];
  Xf[i * 256 + t] = x;
  Xh[i * 256 + t] = (_Float16)x;
  if (t == 0) rs[i] = rv;
}

// 8) delta = softplus(raw[:, :256]) * (1 + 2*rs)   (in place, cols 0..255)
__global__ void delta_kernel(float* __restrict__ raw, const float* __restrict__ rs) {
  long i = blockIdx.x;
  int t = threadIdx.x;
  float v = raw[i * 288 + t];
  float sp = (v > 20.f) ? v : log1pf(expf(v));
  raw[i * 288 + t] = sp * (1.f + 2.f * rs[i]);
}

// ---------------------------------------------------------------------------
// 9) Tree scan. Channel t handled by thread t for the whole 1024-step walk:
//    parent state of channel t was written by the same thread -> no barriers.
// ---------------------------------------------------------------------------
__global__ __launch_bounds__(256) void scan_kernel(const float* __restrict__ raw,
                                                   const float* __restrict__ Xf,
                                                   const int* __restrict__ par,
                                                   const int* __restrict__ bfs,
                                                   const float* __restrict__ A_log,
                                                   const float* __restrict__ Dv,
                                                   float* __restrict__ H,
                                                   float* __restrict__ Y,
                                                   _Float16* __restrict__ Yh) {
  int b = blockIdx.x, t = threadIdx.x;
  float Ar[16];
#pragma unroll
  for (int s = 0; s < 16; ++s) Ar[s] = -expf(A_log[t * 16 + s]);
  float Dt = Dv[t];
  for (int i = 0; i < 1024; ++i) {
    long gi = (long)b * 1024 + i;
    float d = raw[gi * 288 + t];
    float x = Xf[gi * 256 + t];
    int p = par[gi];
    const float* hp = (p >= 0) ? (H + (((long)b * 1024 + p) * 256 + t) * 16) : nullptr;
    float* hw = H + (gi * 256 + t) * 16;
    float y = 0.f;
#pragma unroll
    for (int s = 0; s < 16; ++s) {
      float hpv = hp ? hp[s] : 0.f;
      float Bs = raw[gi * 288 + 256 + s];
      float Cs = raw[gi * 288 + 272 + s];
      float hv = expf(d * Ar[s]) * hpv + d * Bs * x;
      hw[s] = hv;
      y += hv * Cs;
    }
    y += Dt * x;
    int node = bfs[gi];
    long oi = ((long)b * 1024 + node) * 256 + t;  // scatter back to raster order
    Y[oi] = y;
    Yh[oi] = (_Float16)y;
  }
}

// 10) LayerNorm over last dim (256) -> d_out
__global__ __launch_bounds__(256) void layernorm_kernel(const float* __restrict__ o,
                                                        const float* __restrict__ g,
                                                        const float* __restrict__ be,
                                                        float* __restrict__ out) {
  __shared__ float red[256];
  long row = blockIdx.x;
  int t = threadIdx.x;
  float v = o[row * 256 + t];
  red[t] = v; __syncthreads();
  for (int s = 128; s > 0; s >>= 1) { if (t < s) red[t] += red[t + s]; __syncthreads(); }
  float mu = red[0] * (1.f / 256.f);
  __syncthreads();
  float dv = v - mu;
  red[t] = dv * dv; __syncthreads();
  for (int s = 128; s > 0; s >>= 1) { if (t < s) red[t] += red[t + s]; __syncthreads(); }
  float var = red[0] * (1.f / 256.f);
  out[row * 256 + t] = g[t] * dv * rsqrtf(var + 1e-5f) + be[t];
}

// ---------------------------------------------------------------------------
extern "C" void kernel_launch(void* const* d_in, const int* in_sizes, int n_in,
                              void* d_out, int out_size, void* d_ws, size_t ws_size,
                              hipStream_t stream) {
  const float* images  = (const float*)d_in[0];
  const float* lang_g  = (const float*)d_in[1];
  const float* conv1_w = (const float*)d_in[2];
  const float* conv1_b = (const float*)d_in[3];
  const float* conv2_w = (const float*)d_in[4];
  const float* conv2_b = (const float*)d_in[5];
  const float* fc_w    = (const float*)d_in[6];
  const float* fc_b    = (const float*)d_in[7];
  const float* W_gate  = (const float*)d_in[8];
  const float* W_gp    = (const float*)d_in[9];
  const float* A_log   = (const float*)d_in[10];
  const float* Dv      = (const float*)d_in[11];
  const float* B_w     = (const float*)d_in[12];
  const float* B_b     = (const float*)d_in[13];
  const float* C_w     = (const float*)d_in[14];
  const float* C_b     = (const float*)d_in[15];
  const float* Wd_w    = (const float*)d_in[16];
  const float* Wd_b    = (const float*)d_in[17];
  const float* out_w   = (const float*)d_in[18];
  const float* out_b   = (const float*)d_in[19];
  const float* ln_g    = (const float*)d_in[20];
  const float* ln_b    = (const float*)d_in[21];
  float* outp = (float*)d_out;

  // bump allocator over workspace (256B aligned)
  char* ws = (char*)d_ws;
  size_t off = 0;
  auto alloc = [&](size_t bytes) -> char* {
    char* p = ws + off;
    off += (bytes + 255) & ~(size_t)255;
    return p;
  };
  _Float16* w2r     = (_Float16*)alloc(128 * 576 * 2);
  _Float16* fcB     = (_Float16*)alloc(128 * 256 * 2);
  _Float16* WALL    = (_Float16*)alloc(256 * 288 * 2);
  _Float16* WoutB   = (_Float16*)alloc(256 * 256 * 2);
  float*    biasALL = (float*)   alloc(288 * 4);
  _Float16* conv1o  = (_Float16*)alloc((size_t)2048 * 64 * 256 * 2);
  float*    pooled  = (float*)   alloc(2048 * 128 * 4);
  _Float16* pooledh = (_Float16*)alloc(2048 * 128 * 2);
  float*    feats   = (float*)   alloc((size_t)2048 * 256 * 4);
  float*    gproj   = (float*)   alloc(512 * 4);
  float*    gprime  = (float*)   alloc(512 * 4);
  float*    rsem    = (float*)   alloc(2048 * 4);
  int*      bfs     = (int*)     alloc(2048 * 4);
  int*      par     = (int*)     alloc(2048 * 4);
  float*    Xf      = (float*)   alloc((size_t)2048 * 256 * 4);
  _Float16* Xh      = (_Float16*)alloc((size_t)2048 * 256 * 2);
  float*    rs      = (float*)   alloc(2048 * 4);
  float*    raw     = (float*)   alloc((size_t)2048 * 288 * 4);
  float*    H       = (float*)   alloc((size_t)2048 * 256 * 16 * 4);
  float*    Y       = (float*)   alloc((size_t)2048 * 256 * 4);
  _Float16* Yh      = (_Float16*)alloc((size_t)2048 * 256 * 2);
  float*    o       = (float*)   alloc((size_t)2048 * 256 * 4);
  (void)ws_size; (void)in_sizes; (void)n_in; (void)out_size;

  prep_weights_kernel<<<240, 256, 0, stream>>>(conv2_w, fc_w, Wd_w, B_w, C_w, out_w,
                                               Wd_b, B_b, C_b,
                                               w2r, fcB, WALL, WoutB, biasALL);
  conv1_kernel<<<2048, 256, 0, stream>>>(images, conv1_w, conv1_b, conv1o);
  conv2_pool_kernel<<<2048, 512, 0, stream>>>(conv1o, w2r, conv2_b, pooled, pooledh);
  // fc: feats = pooled @ fc_w^T + fc_b      (M=2048, N=256, K=128)
  gemm_f16_kernel<<<dim3(64, 2), 256, 0, stream>>>(pooledh, fcB, fc_b, feats, nullptr,
                                                   2048, 256, 128);
  lang_proj_kernel<<<2, 256, 0, stream>>>(lang_g, W_gate, W_gp, gproj, gprime);
  rsem_kernel<<<8, 256, 0, stream>>>(gproj, feats, rsem);
  structure_kernel<<<2, 256, 0, stream>>>(feats, rsem, bfs, par);
  build_x_kernel<<<2048, 256, 0, stream>>>(feats, rsem, gprime, bfs, Xf, Xh, rs);
  // fused [delta|Bp|Cp] = X @ [Wd|B|C]^T + bias   (M=2048, N=288, K=256)
  gemm_f16_kernel<<<dim3(64, 3), 256, 0, stream>>>(Xh, WALL, biasALL, raw, nullptr,
                                                   2048, 288, 256);
  delta_kernel<<<2048, 256, 0, stream>>>(raw, rs);
  scan_kernel<<<2, 256, 0, stream>>>(raw, Xf, par, bfs, A_log, Dv, H, Y, Yh);
  // out projection: o = Y @ out_w^T + out_b  (M=2048, N=256, K=256)
  gemm_f16_kernel<<<dim3(64, 2), 256, 0, stream>>>(Yh, WoutB, out_b, o, nullptr,
                                                   2048, 256, 256);
  layernorm_kernel<<<2048, 256, 0, stream>>>(o, ln_g, ln_b, outp);
}
